// SpikeMLP_55250459296348
// MI455X (gfx1250) — matrix operand order repaired and checked by reading, not compiled, not run on hardware
//
#include <hip/hip_runtime.h>

// ---------------------------------------------------------------------------
// Fused spiking-MLP for gfx1250 (MI455X), v2.
//  - f16 WMMA, f32 accumulate; cur state = WMMA accumulators in VGPRs.
//  - volt state moved to wave-private LDS (fixes v1 scratch spills: register
//    demand now ~180 < 256 VGPRs).
//  - kt-outer / nt-inner loop: one A fragment feeds all of a wave's N-tiles
//    (4x less LDS A traffic, 4 independent WMMA chains per A fragment).
//  - weights pre-packed to B-fragment order f16 in d_ws (L2-resident stream).
// ---------------------------------------------------------------------------

typedef __attribute__((ext_vector_type(16))) _Float16 v16h;
typedef __attribute__((ext_vector_type(8)))  float    v8f;

#define B_TOTAL 4096
#define D_INP   1024
#define T_STEPS 16
#define NH1     1024
#define NH2     1024
#define NOUT    512

#define BM      16          // batch rows per workgroup (WMMA M)
#define NTHR    512         // 16 waves -> 4 per SIMD, 256-VGPR budget each
#define NWAVE   16
#define NT1     4           // N-tiles per wave, layers 1 & 2 (16*4*16 = 1024)
#define NTO     2           // N-tiles per wave, output layer (16*2*16 = 512)
#define SP      1032        // LDS spike-row stride in halves: 1024 + 8 (16B pad)

#define VTH     0.5f
#define CDECAY  0.5f
#define VDECAY  0.75f

// LDS volt banks, f32, layout [globalTile][h(0..7)][lane(0..31)] -> conflict-free
#define V1_OFF  0                        // 16 waves * 4 tiles * 256 = 16384
#define V2_OFF  16384
#define VO_OFF  32768                    // 16 waves * 2 tiles * 256 = 8192
#define V_TOTAL 40960                    // 160 KB

union F16x16 { v16h v; uint4 q[2]; };

// A fragment (16x32 f16, M x K) from row-major LDS [BM][SP].
__device__ __forceinline__ v16h load_a_frag(const _Float16* buf, int lane, int k0) {
  int off = (lane & 15) * SP + k0 + ((lane >> 4) << 3);
  F16x16 u;
  u.q[0] = *(const uint4*)(buf + off);
  u.q[1] = *(const uint4*)(buf + off + 16);
  return u.v;
}

// B fragment (32x16 f16, K x N): each lane reads its contiguous 32B chunk.
__device__ __forceinline__ v16h load_b_frag(const _Float16* __restrict__ wpk,
                                            int tile, int lane) {
  const uint4* p = (const uint4*)(wpk + (((size_t)tile * 32 + lane) << 4));
  F16x16 u;
  u.q[0] = p[0];
  u.q[1] = p[1];
  return u.v;
}

// Pack row-major fp32 W[Nout][K] into B-fragment-order f16.
__global__ void pack_weights_kernel(const float* __restrict__ W,
                                    _Float16* __restrict__ pk,
                                    int Nout, int K) {
  int idx = blockIdx.x * blockDim.x + threadIdx.x;
  if (idx >= Nout * K) return;
  int i    = idx & 15;
  int lane = (idx >> 4) & 31;
  int tile = idx >> 9;
  int Kt   = K >> 5;
  int tn   = tile / Kt;
  int tk   = tile - tn * Kt;
  int n = tn * 16 + (lane & 15);
  int k = tk * 32 + ((lane & 16) >> 1) + (i & 7) + ((i & 8) << 1);
  pk[idx] = (_Float16)W[(size_t)n * K + k];
}

__global__ __launch_bounds__(NTHR, 1)
void spike_mlp_kernel(const float* __restrict__ xin,     // [B][D_INP][T]
                      const _Float16* __restrict__ w1pk, const float* __restrict__ b1,
                      const _Float16* __restrict__ w2pk, const float* __restrict__ b2,
                      const _Float16* __restrict__ wopk, const float* __restrict__ bo,
                      float* __restrict__ out)           // [B][NOUT]
{
  __shared__ _Float16 bufX[BM * SP];    // layer input (x, then s2)  33 KB
  __shared__ _Float16 bufS[BM * SP];    // s1                        33 KB
  __shared__ float    voltLds[V_TOTAL]; // volt state, wave-private 160 KB

  const int tid   = threadIdx.x;
  const int lane  = tid & 31;
  const int wid   = tid >> 5;
  const int b0    = blockIdx.x * BM;
  const int col16 = lane & 15;          // N within tile (C/D layout)
  const int rowhi = (lane >> 4) << 3;   // M offset 0 or 8 (C/D layout)

  // zero volt state
  for (int i = tid; i < V_TOTAL; i += NTHR) voltLds[i] = 0.f;

  // hoisted per-lane biases
  float bias1[NT1], bias2[NT1], biaso[NTO];
#pragma unroll
  for (int nt = 0; nt < NT1; ++nt) {
    bias1[nt] = b1[(wid * NT1 + nt) * 16 + col16];
    bias2[nt] = b2[(wid * NT1 + nt) * 16 + col16];
  }
#pragma unroll
  for (int nt = 0; nt < NTO; ++nt)
    biaso[nt] = bo[(wid * NTO + nt) * 16 + col16];

  // Persistent cur state == WMMA accumulators (C-fragment layout, VGPRs).
  v8f cur1[NT1] = {}, cur2[NT1] = {};
  v8f curo[NTO] = {}, sumo[NTO] = {};

  __syncthreads();   // volt zero-init visible (wave-private, but cheap & safe)

#pragma unroll 1
  for (int t = 0; t < T_STEPS; ++t) {
    // ---- stage input spikes for this timestep: fp32 (stride T) -> f16 LDS
    for (int i = tid; i < BM * D_INP; i += NTHR) {
      int row = i >> 10, col = i & (D_INP - 1);
      float v = xin[((size_t)(b0 + row) * D_INP + col) * T_STEPS + t];
      bufX[row * SP + col] = (_Float16)v;
    }
    __syncthreads();

    // ======== layer 1: x (bufX) -> s1 (bufS) ========
#pragma unroll
    for (int nt = 0; nt < NT1; ++nt) {
#pragma unroll
      for (int h = 0; h < 8; ++h) cur1[nt][h] = cur1[nt][h] * CDECAY + bias1[nt];
    }
#pragma unroll 4
    for (int kt = 0; kt < 32; ++kt) {
      v16h a = load_a_frag(bufX, lane, kt * 32);
#pragma unroll
      for (int nt = 0; nt < NT1; ++nt) {
        v16h b = load_b_frag(w1pk, (wid * NT1 + nt) * 32 + kt, lane);
        cur1[nt] = __builtin_amdgcn_wmma_f32_16x16x32_f16(false, a, false, b,
                                                          (short)0, cur1[nt], false, false);
      }
    }
#pragma unroll
    for (int nt = 0; nt < NT1; ++nt) {
      int gt   = wid * NT1 + nt;
      int vidx = V1_OFF + gt * 256 + lane;
#pragma unroll
      for (int h = 0; h < 8; ++h) {
        float vo   = voltLds[vidx + h * 32];
        float keep = (vo > VTH) ? 0.f : 1.f;          // (1 - prev spike)
        float vn   = vo * VDECAY * keep + cur1[nt][h];
        voltLds[vidx + h * 32] = vn;
        bufS[(h + rowhi) * SP + gt * 16 + col16] = (_Float16)((vn > VTH) ? 1.f : 0.f);
      }
    }
    __syncthreads();

    // ======== layer 2: s1 (bufS) -> s2 (bufX) ========
#pragma unroll
    for (int nt = 0; nt < NT1; ++nt) {
#pragma unroll
      for (int h = 0; h < 8; ++h) cur2[nt][h] = cur2[nt][h] * CDECAY + bias2[nt];
    }
#pragma unroll 4
    for (int kt = 0; kt < 32; ++kt) {
      v16h a = load_a_frag(bufS, lane, kt * 32);
#pragma unroll
      for (int nt = 0; nt < NT1; ++nt) {
        v16h b = load_b_frag(w2pk, (wid * NT1 + nt) * 32 + kt, lane);
        cur2[nt] = __builtin_amdgcn_wmma_f32_16x16x32_f16(false, a, false, b,
                                                          (short)0, cur2[nt], false, false);
      }
    }
#pragma unroll
    for (int nt = 0; nt < NT1; ++nt) {
      int gt   = wid * NT1 + nt;
      int vidx = V2_OFF + gt * 256 + lane;
#pragma unroll
      for (int h = 0; h < 8; ++h) {
        float vo   = voltLds[vidx + h * 32];
        float keep = (vo > VTH) ? 0.f : 1.f;
        float vn   = vo * VDECAY * keep + cur2[nt][h];
        voltLds[vidx + h * 32] = vn;
        bufX[(h + rowhi) * SP + gt * 16 + col16] = (_Float16)((vn > VTH) ? 1.f : 0.f);
      }
    }
    __syncthreads();

    // ======== output layer: s2 (bufX) -> spike-count accumulator ========
#pragma unroll
    for (int nt = 0; nt < NTO; ++nt) {
#pragma unroll
      for (int h = 0; h < 8; ++h) curo[nt][h] = curo[nt][h] * CDECAY + biaso[nt];
    }
#pragma unroll 4
    for (int kt = 0; kt < 32; ++kt) {
      v16h a = load_a_frag(bufX, lane, kt * 32);
#pragma unroll
      for (int nt = 0; nt < NTO; ++nt) {
        v16h b = load_b_frag(wopk, (wid * NTO + nt) * 32 + kt, lane);
        curo[nt] = __builtin_amdgcn_wmma_f32_16x16x32_f16(false, a, false, b,
                                                          (short)0, curo[nt], false, false);
      }
    }
#pragma unroll
    for (int nt = 0; nt < NTO; ++nt) {
      int gt   = wid * NTO + nt;
      int vidx = VO_OFF + gt * 256 + lane;
#pragma unroll
      for (int h = 0; h < 8; ++h) {
        float vo   = voltLds[vidx + h * 32];
        float keep = (vo > VTH) ? 0.f : 1.f;
        float vn   = vo * VDECAY * keep + curo[nt][h];
        voltLds[vidx + h * 32] = vn;
        sumo[nt][h] += (vn > VTH) ? 1.f : 0.f;
      }
    }
    __syncthreads();   // protect bufX before next timestep's x-fill
  }

  // ---- write mean output spikes: C/D layout -> row-major [B][NOUT]
#pragma unroll
  for (int nt = 0; nt < NTO; ++nt) {
    int n = (wid * NTO + nt) * 16 + col16;
#pragma unroll
    for (int h = 0; h < 8; ++h) {
      out[(size_t)(b0 + h + rowhi) * NOUT + n] = sumo[nt][h] * (1.0f / T_STEPS);
    }
  }
}

extern "C" void kernel_launch(void* const* d_in, const int* in_sizes, int n_in,
                              void* d_out, int out_size, void* d_ws, size_t ws_size,
                              hipStream_t stream) {
  const float* xin = (const float*)d_in[0];
  const float* W1  = (const float*)d_in[1];
  const float* b1  = (const float*)d_in[2];
  const float* W2  = (const float*)d_in[3];
  const float* b2  = (const float*)d_in[4];
  const float* Wo  = (const float*)d_in[5];
  const float* bo  = (const float*)d_in[6];
  (void)in_sizes; (void)n_in; (void)out_size; (void)ws_size;

  // Workspace: packed f16 weights (5 MB total).
  _Float16* w1pk = (_Float16*)d_ws;
  _Float16* w2pk = w1pk + (size_t)NH1 * D_INP;
  _Float16* wopk = w2pk + (size_t)NH2 * NH1;

  pack_weights_kernel<<<(NH1 * D_INP + 255) / 256, 256, 0, stream>>>(W1, w1pk, NH1, D_INP);
  pack_weights_kernel<<<(NH2 * NH1 + 255) / 256, 256, 0, stream>>>(W2, w2pk, NH2, NH1);
  pack_weights_kernel<<<(NOUT * NH2 + 255) / 256, 256, 0, stream>>>(Wo, wopk, NOUT, NH2);

  spike_mlp_kernel<<<B_TOTAL / BM, NTHR, 0, stream>>>(
      xin, w1pk, b1, w2pk, b2, wopk, bo, (float*)d_out);
}
